// GRU_50663434223726
// MI455X (gfx1250) — matrix operand order repaired and checked by reading, not compiled, not run on hardware
//
#include <hip/hip_runtime.h>
#include <hip/hip_bf16.h>
#include <math.h>

// ---------------------------------------------------------------------------
// GRU scan for MI455X (gfx1250, wave32, WMMA).
//
// Phase 0: pack W[:, :2048] (recurrent) and W[:, 2048:] (input) into f16
//          WMMA A-fragment layout; pack inputs into B-fragment layout.
// Phase 1: X[g][t][j] = W_g[:,2048:] @ x_t + b_g  (WMMA GEMM, f16 in/f32 acc,
//          bias folded so the scan never touches it).
// Phase 2: persistent 32-block kernel scans t=0..8191. Recurrent matvecs use
//          WMMA with the h / (r*h) vector in column 0 of the B operand.
//          Per-step critical path minimized: 4-way accumulator ILP (z/r),
//          2-way split-K across wave pairs + LDS reduce (h~). Grid barriers
//          via monotone ticket counter in workspace. Weights stay L2-resident
//          (25 MB f16); streaming X/out traffic uses non-temporal hints.
//
// Workspace requirement: ~155 MB.
// ---------------------------------------------------------------------------

typedef __attribute__((ext_vector_type(16))) _Float16 v16h;
typedef __attribute__((ext_vector_type(8)))  float    v8f;

#define SEQ_LEN 8192
#define IN_SZ   1024
#define HID     2048
#define CONCAT  3072
#define JT      (HID/16)      // 128 row tiles of 16
#define KC_REC  (HID/32)      // 64  k-chunks of 32 (recurrent part)
#define KC_X    (IN_SZ/32)    // 32  k-chunks of 32 (input part)
#define TT      (SEQ_LEN/16)  // 512 time tiles of 16
#define NB      32            // persistent blocks

// A-fragment (16x32 f16, MxK) lane/half -> K mapping: k = (lane/16)*16 + half
__device__ __forceinline__ int a_k(int lane, int h) {
    return ((lane >= 16) ? 16 : 0) + h;
}

// One WMMA matvec step: A tile chunk c, vector chunk c in column 0 of B.
__device__ __forceinline__ v8f mv_step(const _Float16* __restrict__ Ab,
                                       const _Float16* __restrict__ vec,
                                       int c, int lane, v8f acc) {
    v16h a = *(const v16h*)(Ab + c * 512 + lane * 16);
    v16h bf = {};
    if ((lane & 15) == 0)
        bf = *(const v16h*)(vec + c * 32 + (lane >> 4) * 16);
    return __builtin_amdgcn_wmma_f32_16x16x32_f16(
        false, a, false, bf, (short)0, acc, false, false);
}

// ---------------------------------------------------------------------------
// Packing kernels
// ---------------------------------------------------------------------------
__global__ void pack_A_rec(const float* Wz, const float* Wr, const float* Wh,
                           _Float16* dst) {
    long idx = (long)blockIdx.x * blockDim.x + threadIdx.x;
    if (idx >= 3L * JT * KC_REC * 512) return;
    int half = idx & 15;
    int lane = (int)((idx >> 4) & 31);
    long tile = idx >> 9;
    int c  = (int)(tile & (KC_REC - 1));
    long t2 = tile >> 6;
    int jt = (int)(t2 & (JT - 1));
    int g  = (int)(t2 >> 7);
    int m = lane & 15;
    int k = a_k(lane, half);
    const float* W = (g == 0) ? Wz : ((g == 1) ? Wr : Wh);
    dst[idx] = (_Float16)W[(long)(jt * 16 + m) * CONCAT + c * 32 + k];
}

__global__ void pack_A_x(const float* Wz, const float* Wr, const float* Wh,
                         _Float16* dst) {
    long idx = (long)blockIdx.x * blockDim.x + threadIdx.x;
    if (idx >= 3L * JT * KC_X * 512) return;
    int half = idx & 15;
    int lane = (int)((idx >> 4) & 31);
    long tile = idx >> 9;
    int c  = (int)(tile & (KC_X - 1));
    long t2 = tile >> 5;
    int jt = (int)(t2 & (JT - 1));
    int g  = (int)(t2 >> 7);
    int m = lane & 15;
    int k = a_k(lane, half);
    const float* W = (g == 0) ? Wz : ((g == 1) ? Wr : Wh);
    dst[idx] = (_Float16)W[(long)(jt * 16 + m) * CONCAT + HID + c * 32 + k];
}

// B-fragment (32x16 f16, KxN): lane = (k/16)*16 + n, half = k%16
__global__ void pack_B_in(const float* inputs, _Float16* dst) {
    long idx = (long)blockIdx.x * blockDim.x + threadIdx.x;
    if (idx >= (long)TT * KC_X * 512) return;
    int half = idx & 15;
    int lane = (int)((idx >> 4) & 31);
    long tile = idx >> 9;
    int c  = (int)(tile & (KC_X - 1));
    int tt = (int)(tile >> 5);
    int n = lane & 15;
    int k = (lane >> 4) * 16 + half;
    dst[idx] = (_Float16)inputs[(long)(tt * 16 + n) * IN_SZ + c * 32 + k];
}

// ---------------------------------------------------------------------------
// Phase 1: X[g][t][j] = W_g[:, 2048:] @ x_t + b_g   (one 16x16 tile per wave)
// ---------------------------------------------------------------------------
__global__ __launch_bounds__(256) void xproj(const _Float16* __restrict__ Ax,
                                             const _Float16* __restrict__ Bin,
                                             const float* __restrict__ bz,
                                             const float* __restrict__ br,
                                             const float* __restrict__ bh,
                                             _Float16* __restrict__ X) {
    long gwave = ((long)blockIdx.x * blockDim.x + threadIdx.x) >> 5;
    int lane = threadIdx.x & 31;
    if (gwave >= 3L * JT * TT) return;
    int tt = (int)(gwave & (TT - 1));
    long r = gwave >> 9;
    int jt = (int)(r & (JT - 1));
    int g  = (int)(r >> 7);

    const _Float16* Ab = Ax + ((long)(g * JT + jt) * KC_X) * 512;
    const _Float16* Bb = Bin + ((long)tt * KC_X) * 512;
    v8f acc0 = {}, acc1 = {};
    for (int c = 0; c < KC_X; c += 2) {
        v16h a0 = *(const v16h*)(Ab + c * 512 + lane * 16);
        v16h b0 = *(const v16h*)(Bb + c * 512 + lane * 16);
        acc0 = __builtin_amdgcn_wmma_f32_16x16x32_f16(
            false, a0, false, b0, (short)0, acc0, false, false);
        v16h a1 = *(const v16h*)(Ab + (c + 1) * 512 + lane * 16);
        v16h b1 = *(const v16h*)(Bb + (c + 1) * 512 + lane * 16);
        acc1 = __builtin_amdgcn_wmma_f32_16x16x32_f16(
            false, a1, false, b1, (short)0, acc1, false, false);
    }
    v8f acc = acc0 + acc1;
    // D: lane -> col n = lane%16 (time), rows m = (lane/16)*8 + v (hidden j)
    int n = lane & 15;
    int mb = (lane >> 4) * 8;
    const float* bias = (g == 0) ? bz : ((g == 1) ? br : bh);
    long base = ((long)g * SEQ_LEN + tt * 16 + n) * HID + jt * 16 + mb;
    for (int v = 0; v < 8; ++v)
        X[base + v] = (_Float16)(acc[v] + bias[jt * 16 + mb + v]);
}

// ---------------------------------------------------------------------------
// Grid barrier (monotone ticket, deterministic, graph-capture safe)
// ---------------------------------------------------------------------------
__device__ __forceinline__ void grid_barrier(volatile unsigned* cnt) {
    __syncthreads();
    if (threadIdx.x == 0) {
        __threadfence();
        unsigned ticket = atomicAdd((unsigned*)cnt, 1u);
        unsigned target = (ticket / NB + 1u) * NB;
        while (*cnt < target) { __builtin_amdgcn_s_sleep(1); }
        __threadfence();
    }
    __syncthreads();
}

// ---------------------------------------------------------------------------
// Phase 2: persistent recurrent scan. Block b owns hidden rows [b*64, b*64+64).
// z/r phase: 8 waves = 2 gates x 4 tiles, 4 accumulators each (chain depth 16).
// h~ phase: 8 waves = 4 tiles x 2-way split-K, LDS-reduced.
// ---------------------------------------------------------------------------
__global__ __launch_bounds__(256) void gru_scan(
    const _Float16* __restrict__ Arec, const _Float16* __restrict__ X,
    float* __restrict__ out, float* h32, _Float16* h16, _Float16* rh16,
    unsigned* bar) {
    __shared__ float z_s[64];
    __shared__ float part[4][16];  // split-K partials for h~ (upper half-K)
    int wave = threadIdx.x >> 5;
    int lane = threadIdx.x & 31;
    int b = blockIdx.x;
    int jbase = b * 64;

    for (int t = 0; t < SEQ_LEN; ++t) {
        // ---- z and r gates: wave w -> gate w>>2, tile w&3 ----
        {
            int g  = wave >> 2;          // 0 = z, 1 = r
            int jt = b * 4 + (wave & 3); // global 16-row tile
            const _Float16* Ab = Arec + ((long)(g * JT + jt) * KC_REC) * 512;
            v8f a0 = {}, a1 = {}, a2 = {}, a3 = {};
            for (int c = 0; c < KC_REC; c += 4) {
                a0 = mv_step(Ab, h16, c + 0, lane, a0);
                a1 = mv_step(Ab, h16, c + 1, lane, a1);
                a2 = mv_step(Ab, h16, c + 2, lane, a2);
                a3 = mv_step(Ab, h16, c + 3, lane, a3);
            }
            v8f acc = (a0 + a1) + (a2 + a3);
            if ((lane & 15) == 0) {      // lanes 0 (rows 0-7), 16 (rows 8-15)
                int mb = (lane >> 4) * 8;
                const _Float16* Xg = X + ((long)g * SEQ_LEN + t) * HID;
                for (int v = 0; v < 8; ++v) {
                    int j = jt * 16 + mb + v;
                    float a = acc[v] + (float)__builtin_nontemporal_load(Xg + j);
                    float s = 1.0f / (1.0f + expf(-a));
                    if (g == 0) z_s[j - jbase] = s;
                    else        rh16[j] = (_Float16)(s * h32[j]);
                }
            }
        }
        grid_barrier(bar);  // full r*h vector + z_s visible

        // ---- h~ : wave w -> tile w&3, K-half w>>2 ----
        {
            int tl = wave & 3;
            int kh = wave >> 2;          // 0: chunks 0-31, 1: chunks 32-63
            int jt = b * 4 + tl;
            const _Float16* Ab = Arec + ((long)(2 * JT + jt) * KC_REC) * 512;
            v8f a0 = {}, a1 = {};
            for (int c = kh * 32; c < kh * 32 + 32; c += 2) {
                a0 = mv_step(Ab, rh16, c + 0, lane, a0);
                a1 = mv_step(Ab, rh16, c + 1, lane, a1);
            }
            v8f acc = a0 + a1;
            int mb = (lane >> 4) * 8;
            if (kh == 1 && (lane & 15) == 0)
                for (int v = 0; v < 8; ++v) part[tl][mb + v] = acc[v];
            __syncthreads();
            if (kh == 0 && (lane & 15) == 0) {
                const _Float16* Xh = X + ((long)2 * SEQ_LEN + t) * HID;
                for (int v = 0; v < 8; ++v) {
                    int j = jt * 16 + mb + v;
                    float a = acc[v] + part[tl][mb + v] +
                              (float)__builtin_nontemporal_load(Xh + j);
                    float ht = tanhf(a);
                    float z  = z_s[j - jbase];
                    float hp = h32[j];
                    float hn = (1.0f - z) * hp + z * ht;
                    __builtin_nontemporal_store(hn, out + (long)t * HID + j);
                    h32[j] = hn;
                    h16[j] = (_Float16)hn;
                    if (t == SEQ_LEN - 1) out[(long)SEQ_LEN * HID + j] = hn;
                }
            }
        }
        grid_barrier(bar);  // new h visible before next step's z/r
    }
}

// ---------------------------------------------------------------------------
__global__ void gru_init(unsigned* bar, float* h32, _Float16* h16,
                         _Float16* rh16) {
    int i = blockIdx.x * blockDim.x + threadIdx.x;
    if (i < 64) bar[i] = 0u;
    if (i < HID) {
        h32[i] = 0.0f;
        h16[i] = (_Float16)0.0f;
        rh16[i] = (_Float16)0.0f;
    }
}

extern "C" void kernel_launch(void* const* d_in, const int* in_sizes, int n_in,
                              void* d_out, int out_size, void* d_ws,
                              size_t ws_size, hipStream_t stream) {
    const float* inputs = (const float*)d_in[0];
    const float* Wz = (const float*)d_in[1];
    const float* bz = (const float*)d_in[2];
    const float* Wr = (const float*)d_in[3];
    const float* br = (const float*)d_in[4];
    const float* Wh = (const float*)d_in[5];
    const float* bh = (const float*)d_in[6];
    float* out = (float*)d_out;

    char* ws = (char*)d_ws;
    size_t off = 0;
    auto carve = [&](size_t bytes) -> char* {
        char* p = ws + off;
        off += (bytes + 255) & ~(size_t)255;
        return p;
    };
    unsigned* bar  = (unsigned*)carve(256);
    _Float16* Arec = (_Float16*)carve((size_t)3 * JT * KC_REC * 512 * 2);
    _Float16* Ax   = (_Float16*)carve((size_t)3 * JT * KC_X * 512 * 2);
    _Float16* Bin  = (_Float16*)carve((size_t)TT * KC_X * 512 * 2);
    _Float16* X    = (_Float16*)carve((size_t)3 * SEQ_LEN * HID * 2);
    float*    h32  = (float*)carve((size_t)HID * 4);
    _Float16* h16  = (_Float16*)carve((size_t)HID * 2);
    _Float16* rh16 = (_Float16*)carve((size_t)HID * 2);

    gru_init<<<(HID + 255) / 256, 256, 0, stream>>>(bar, h32, h16, rh16);

    long nrec = 3L * JT * KC_REC * 512;
    pack_A_rec<<<(unsigned)((nrec + 255) / 256), 256, 0, stream>>>(Wz, Wr, Wh, Arec);
    long nax = 3L * JT * KC_X * 512;
    pack_A_x<<<(unsigned)((nax + 255) / 256), 256, 0, stream>>>(Wz, Wr, Wh, Ax);
    long nbin = (long)TT * KC_X * 512;
    pack_B_in<<<(unsigned)((nbin + 255) / 256), 256, 0, stream>>>(inputs, Bin);

    long jobs = 3L * JT * TT;  // one wave per (gate, row-tile, time-tile)
    xproj<<<(unsigned)((jobs * 32 + 255) / 256), 256, 0, stream>>>(
        Ax, Bin, bz, br, bh, X);

    gru_scan<<<NB, 256, 0, stream>>>(Arec, X, out, h32, h16, rh16, bar);
}
